// ScaledDotProductAttention_7980049236476
// MI455X (gfx1250) — compile-verified, matrix operand order checked by reading, not run on hardware
//
#include <hip/hip_runtime.h>

// Neighborhood attention (5x5 window), B=8 C=128 H=W=128, fp32 in/out.
// HBM-bound (~268 MB compulsory @ 23.3 TB/s -> ~12 us floor). Strategy:
// f32->bf16 once into LDS, v_wmma_f32_16x16x32_bf16 with f32 accumulation.
// Phase 1: scores[16 pix x 20 halo cols] = Q[16x128] x K[128x20] per dy.
// Phase 2: out[16 pix x 128 ch] += attnBand[16x32] x V[32x16] per dy/chunk.

#define BATCH 8
#define CHN   128
#define HH    128
#define WW    128
#define TB    8      // rows per block (one wave per row)
#define TW    16     // cols per block (WMMA M)
#define HALO_R (TB + 4)   // 12
#define HALO_C 32         // 20 used, padded+zero-filled to 32 (WMMA N/K)
#define NTHREADS 256

typedef __attribute__((ext_vector_type(16))) __bf16        v16bf;
typedef __attribute__((ext_vector_type(8)))  float         v8f;
typedef __attribute__((ext_vector_type(4)))  unsigned int  u32x4;

union Frag {
    u32x4          q[2];   // two 16B quarters -> (VGPR0..3, VGPR4..7)
    v16bf          v;
    unsigned short us[16];
};

__device__ __forceinline__ unsigned short f2bf(float f) {
    union { float f; unsigned u; } x; x.f = f;
    unsigned r = x.u + 0x7FFFu + ((x.u >> 16) & 1u);   // round-to-nearest-even
    return (unsigned short)(r >> 16);
}

__global__ __launch_bounds__(NTHREADS)
void natt5x5_kernel(const float* __restrict__ Q,
                    const float* __restrict__ K,
                    const float* __restrict__ V,
                    float* __restrict__ O)
{
    // LDS tiles (bf16). Zero-filled halo exactly matches jnp.pad semantics
    // (zero scores DO participate in the softmax, as in the reference).
    __shared__ unsigned short qtile[TB * TW * CHN];           // [r][m][c]      32 KB
    __shared__ unsigned short ktile[HALO_R * HALO_C * CHN];   // [row][col][c]  96 KB
    __shared__ unsigned short vtile[HALO_R * CHN * HALO_C];   // [row][c][col]  96 KB
    __shared__ float          scbuf[TB * 2048];               // scores / out staging 64 KB
    __shared__ unsigned short attnbuf[TB * 5 * 16 * 8];       // attn weights 10 KB

    const int tid = threadIdx.x;
    const int b  = blockIdx.z;
    const int h0 = blockIdx.y * TB;
    const int w0 = blockIdx.x * TW;

    // ---------------- cooperative tile load (f32 -> bf16) ----------------
    // Q: [r][m][c], m fastest across lanes -> coalesced 128B/wave reads
    for (int i = tid; i < TB * TW * CHN; i += NTHREADS) {
        int m = i & 15, c = (i >> 4) & (CHN - 1), r = i >> 11;
        float val = Q[(((b * CHN + c) * HH) + (h0 + r)) * WW + (w0 + m)];
        qtile[(r * TW + m) * CHN + c] = f2bf(val);
    }
    // K halo: [row][col][c]; cols 20..31 and out-of-image -> 0
    for (int i = tid; i < HALO_R * HALO_C * CHN; i += NTHREADS) {
        int mc = i & 31, c = (i >> 5) & (CHN - 1), row = i >> 12;
        int gh = h0 - 2 + row, gw = w0 - 2 + mc;
        float val = 0.0f;
        if (mc < 20 && (unsigned)gh < (unsigned)HH && (unsigned)gw < (unsigned)WW)
            val = K[(((b * CHN + c) * HH) + gh) * WW + gw];
        ktile[(row * HALO_C + mc) * CHN + c] = f2bf(val);
    }
    // V halo, LDS-transposed: [row][c][col] so phase-2 B fragments are contiguous
    for (int i = tid; i < HALO_R * HALO_C * CHN; i += NTHREADS) {
        int mc = i & 31, c = (i >> 5) & (CHN - 1), row = i >> 12;
        int gh = h0 - 2 + row, gw = w0 - 2 + mc;
        float val = 0.0f;
        if (mc < 20 && (unsigned)gh < (unsigned)HH && (unsigned)gw < (unsigned)WW)
            val = V[(((b * CHN + c) * HH) + gh) * WW + gw];
        vtile[(row * CHN + c) * HALO_C + mc] = f2bf(val);
    }
    __syncthreads();

    const int wave = tid >> 5, lane = tid & 31;
    const int lane16 = lane & 15, half = lane >> 4;
    const int r = wave;                           // output row within tile
    float*          scw  = &scbuf[wave * 2048];   // [5 dy][16 m][20 n]
    unsigned short* attw = &attnbuf[wave * 640];  // [5 dy][16 m][8 dx]

    // Q fragments are dy-invariant: load all 4 channel-chunks once.
    Frag fq[4];
#pragma unroll
    for (int kc = 0; kc < 4; ++kc) {
        const u32x4* qp = (const u32x4*)&qtile[(r * TW + lane16) * CHN + kc * 32];
        fq[kc].q[0] = qp[half]; fq[kc].q[1] = qp[2 + half];
    }

    // ---------------- phase 1: scores = Q x K^T (banded) ----------------
#pragma unroll
    for (int dy = 0; dy < 5; ++dy) {
        v8f acc0 = {}; v8f acc1 = {};
        const int krow = r + dy;                  // halo row index
#pragma unroll
        for (int kc = 0; kc < 4; ++kc) {          // 128 channels in K=32 chunks
            Frag fb;                              // B = K[32 ch x 16 cols], cols 0..15
            const u32x4* kp0 = (const u32x4*)&ktile[(krow * HALO_C + lane16) * CHN + kc * 32];
            fb.q[0] = kp0[half]; fb.q[1] = kp0[2 + half];
            acc0 = __builtin_amdgcn_wmma_f32_16x16x32_bf16(
                       false, fq[kc].v, false, fb.v, (short)0, acc0, false, false);
            const u32x4* kp1 = (const u32x4*)&ktile[(krow * HALO_C + 16 + lane16) * CHN + kc * 32];
            fb.q[0] = kp1[half]; fb.q[1] = kp1[2 + half];   // cols 16..31 (>=20 are zeros)
            acc1 = __builtin_amdgcn_wmma_f32_16x16x32_bf16(
                       false, fq[kc].v, false, fb.v, (short)0, acc1, false, false);
        }
        // spill score tiles to per-wave LDS: D layout M=r8+8*half, N=lane16
#pragma unroll
        for (int r8 = 0; r8 < 8; ++r8) {
            int M = r8 + 8 * half;
            scw[(dy * 16 + M) * 20 + lane16] = acc0[r8];
            int n1 = 16 + lane16;
            if (n1 < 20) scw[(dy * 16 + M) * 20 + n1] = acc1[r8];
        }
    }
    asm volatile("s_wait_dscnt 0" ::: "memory");

    // ---------------- softmax over the 25-window (lane m = pixel) -------
    if (lane < 16) {
        const int m = lane;
        float vals[25];
        float mx = -3.4e38f;
#pragma unroll
        for (int dy = 0; dy < 5; ++dy)
#pragma unroll
            for (int dx = 0; dx < 5; ++dx) {
                float s = scw[(dy * 16 + m) * 20 + (m + dx)] * 0.08838834764831845f;
                vals[dy * 5 + dx] = s;
                mx = fmaxf(mx, s);
            }
        float sum = 0.0f;
#pragma unroll
        for (int p = 0; p < 25; ++p) { float e = __expf(vals[p] - mx); vals[p] = e; sum += e; }
        float inv = 1.0f / sum;
#pragma unroll
        for (int dy = 0; dy < 5; ++dy)
#pragma unroll
            for (int dx = 0; dx < 5; ++dx)
                attw[(dy * 16 + m) * 8 + dx] = f2bf(vals[dy * 5 + dx] * inv);
    }
    asm volatile("s_wait_dscnt 0" ::: "memory");

    // ---------------- phase 2: out = attn(banded) x V -------------------
    v8f o[8];
#pragma unroll
    for (int cc = 0; cc < 8; ++cc) o[cc] = (v8f){};
#pragma unroll
    for (int dy = 0; dy < 5; ++dy) {
        const int vrow = r + dy;
        Frag fa;                                  // A = banded attn [16 pix x 32 halo-cols]
#pragma unroll
        for (int e = 0; e < 16; ++e) {
            int k  = e + 8 * half + ((e >= 8) ? 8 : 0);   // K index per ISA A layout
            int dx = k - lane16;
            fa.us[e] = (dx >= 0 && dx < 5)
                           ? attw[(dy * 16 + lane16) * 8 + dx]
                           : (unsigned short)0;
        }
#pragma unroll
        for (int cc = 0; cc < 8; ++cc) {          // 128 channels in N=16 chunks
            Frag fb;                              // B = V[32 halo-cols x 16 ch]
            const u32x4* vp = (const u32x4*)&vtile[(vrow * CHN + (cc * 16 + lane16)) * HALO_C];
            fb.q[0] = vp[half]; fb.q[1] = vp[2 + half];
            o[cc] = __builtin_amdgcn_wmma_f32_16x16x32_bf16(
                        false, fa.v, false, fb.v, (short)0, o[cc], false, false);
        }
    }

    // ---------------- transpose through LDS, coalesced store ------------
    float* stg = scw;                             // reuse per-wave scratch: [128 c][16 m]
    asm volatile("s_wait_dscnt 0" ::: "memory");
#pragma unroll
    for (int cc = 0; cc < 8; ++cc)
#pragma unroll
        for (int r8 = 0; r8 < 8; ++r8)
            stg[(cc * 16 + lane16) * 16 + (r8 + 8 * half)] = o[cc][r8];
    asm volatile("s_wait_dscnt 0" ::: "memory");

    const int h = h0 + r;
    for (int i = lane; i < CHN * TW; i += 32) {
        int c = i >> 4, m = i & 15;
        O[(((b * CHN + c) * HH) + h) * WW + (w0 + m)] = stg[c * 16 + m];
    }
}

extern "C" void kernel_launch(void* const* d_in, const int* in_sizes, int n_in,
                              void* d_out, int out_size, void* d_ws, size_t ws_size,
                              hipStream_t stream) {
    const float* q = (const float*)d_in[0];
    const float* k = (const float*)d_in[1];
    const float* v = (const float*)d_in[2];
    float* o = (float*)d_out;
    (void)in_sizes; (void)n_in; (void)out_size; (void)d_ws; (void)ws_size;

    dim3 grid(WW / TW, HH / TB, BATCH);   // (8, 16, 8) = 1024 workgroups
    natt5x5_kernel<<<grid, NTHREADS, 0, stream>>>(q, k, v, o);
}
